// DeformableConv2d_31224412242790
// MI455X (gfx1250) — matrix operand order repaired and checked by reading, not compiled, not run on hardware
//
#include <hip/hip_runtime.h>
#include <math.h>

// ---- types ----
typedef __attribute__((ext_vector_type(16))) _Float16 v16h;
typedef __attribute__((ext_vector_type(8)))  _Float16 h8;
typedef __attribute__((ext_vector_type(2)))  _Float16 h2;
typedef __attribute__((ext_vector_type(8)))  float    v8f;
typedef __attribute__((ext_vector_type(4)))  unsigned int u4;

#define Bn   8
#define Cn   64
#define Hn   128
#define Wn   128
#define On   64
#define KTAP 9
#define KD   576      // 9 taps * 64 channels
#define KSTEPS 18     // KD / 32
#define TILE 64       // pixels per workgroup (half of one W-row: b,y uniform)
#define PROW 584      // LDS row stride in halfs (576 + 8 pad)

union AV { h8 h[2]; v16h v; };

// A-fragment loader using contiguous k order (same order as the B packing):
// lane (m,hi): halfs 0..15 = K(t*32 + 16*hi ..) contiguous
__device__ __forceinline__ AV load_afrag(const _Float16* row, int t, int hi) {
  AV a;
  a.h[0] = *(const h8*)(row + t * 32 + 16 * hi);
  a.h[1] = *(const h8*)(row + t * 32 + 16 * hi + 8);
  return a;
}

// ---------------------------------------------------------------------------
// Prep 1: NCHW f32 -> NHWC f16 (tiled transpose so sampling reads 128B rows)
// ---------------------------------------------------------------------------
__global__ __launch_bounds__(256) void nhwc_kernel(const float* __restrict__ x,
                                                   _Float16* __restrict__ xh) {
  __shared__ float tl[32][33];
  int bid = blockIdx.x;
  int t_hw = bid & 511;          // HW/32 = 512
  int t_c  = (bid >> 9) & 1;     // Cn/32 = 2
  int b    = bid >> 10;
  int tid  = threadIdx.x;
  #pragma unroll
  for (int q = 0; q < 4; ++q) {
    int idx = tid + q * 256;
    int r = idx >> 5;            // c offset (read coalesced along hw)
    int col = idx & 31;
    tl[r][col] = x[((b * Cn + t_c * 32 + r) << 14) + t_hw * 32 + col];
  }
  __syncthreads();
  #pragma unroll
  for (int q = 0; q < 4; ++q) {
    int idx = tid + q * 256;
    int hwl = idx >> 5;
    int cl  = idx & 31;          // write coalesced along c
    xh[(((size_t)(b << 14) + t_hw * 32 + hwl) << 6) + t_c * 32 + cl] =
        (_Float16)tl[cl][hwl];
  }
}

// ---------------------------------------------------------------------------
// Prep 2: pack offset+modulator weights into WMMA-B lane order, f16
//   k = t*32 + 16*(lane>>4) + h  (K index = tap*64 + c)
//   n = ntile*16 + (lane&15)     (n<18: w_off, n<27: w_mod, else 0)
// ---------------------------------------------------------------------------
__global__ void pack_wom_kernel(const float* __restrict__ w_off,
                                const float* __restrict__ w_mod,
                                _Float16* __restrict__ wom) {
  int g = blockIdx.x * blockDim.x + threadIdx.x;
  if (g >= KSTEPS * 2 * 32 * 16) return;
  int h  = g & 15;
  int l  = (g >> 4) & 31;
  int nt = (g >> 9) & 1;
  int t  = g >> 10;
  int k  = t * 32 + ((l >> 4) << 4) + h;
  int n  = nt * 16 + (l & 15);
  int tap = k >> 6, c = k & 63;
  int ky = tap / 3, kx = tap % 3;
  float v = 0.0f;
  if (n < 18)      v = w_off[((n * Cn + c) * 3 + ky) * 3 + kx];
  else if (n < 27) v = w_mod[(((n - 18) * Cn + c) * 3 + ky) * 3 + kx];
  wom[g] = (_Float16)v;
}

// Prep 3: pack w_reg -> [t][ntile=0..3][lane][h], n = output channel o
__global__ void pack_wreg_kernel(const float* __restrict__ w_reg,
                                 _Float16* __restrict__ wrg) {
  int g = blockIdx.x * blockDim.x + threadIdx.x;
  if (g >= KSTEPS * 4 * 32 * 16) return;
  int h  = g & 15;
  int l  = (g >> 4) & 31;
  int nt = (g >> 9) & 3;
  int t  = g >> 11;
  int k  = t * 32 + ((l >> 4) << 4) + h;
  int o  = nt * 16 + (l & 15);
  int tap = k >> 6, c = k & 63;
  wrg[g] = (_Float16)w_reg[((o * Cn + c) * 3 + tap / 3) * 3 + tap % 3];
}

// ---------------------------------------------------------------------------
// Fused main kernel: im2col -> WMMA offset/mod conv -> bilinear sample ->
// WMMA output GEMM.  256 threads (8 wave32), 64 pixels per block.
// ---------------------------------------------------------------------------
__global__ __launch_bounds__(256)
void dcn_main_kernel(const _Float16* __restrict__ xh,
                     const _Float16* __restrict__ wom,
                     const _Float16* __restrict__ wrg,
                     const float* __restrict__ b_off,
                     const float* __restrict__ b_mod,
                     float* __restrict__ out) {
  __shared__ _Float16 P[TILE * PROW];   // im2col tile, then sampled tile
  __shared__ float OM[TILE][28];        // dy0,dx0..dy8,dx8, mod0..mod8

  const int tid  = threadIdx.x;
  const int lane = tid & 31;
  const int wave = tid >> 5;
  const int m    = lane & 15;
  const int hi   = lane >> 4;

  const int p_base = blockIdx.x * TILE;
  const int b   = p_base >> 14;
  const int rem = p_base & 16383;
  const int y   = rem >> 7;
  const int xb  = rem & 127;

  // ---- Stage 1: im2col patch into P (zero-padded borders) ----
  for (int task = tid; task < TILE * KTAP; task += 256) {
    int i = task / KTAP, tap = task % KTAP;
    int yy = y + tap / 3 - 1;
    int xx = xb + i + tap % 3 - 1;
    u4* dst = (u4*)&P[i * PROW + tap * 64];
    if (yy >= 0 && yy < Hn && xx >= 0 && xx < Wn) {
      const u4* src = (const u4*)(xh + (((size_t)(b * Hn + yy) * Wn + xx) << 6));
      #pragma unroll
      for (int q = 0; q < 8; ++q) dst[q] = src[q];
    } else {
      u4 z = {0u, 0u, 0u, 0u};
      #pragma unroll
      for (int q = 0; q < 8; ++q) dst[q] = z;
    }
  }
  __syncthreads();

  // ---- Stage 2: offset/mod GEMM (M=64,N=32,K=576), 4 mt x 2 nt = 8 waves --
  // K-split into two interleaved accumulators (k-steps 0..8 and 9..17) so
  // consecutive WMMAs are independent; summed at the end.
  {
    const int mt = wave >> 1;           // 0..3
    const int nt = wave & 1;            // 0..1
    const int ch = nt * 16 + m;
    float bias = 0.0f;
    if (ch < 18)      bias = b_off[ch];
    else if (ch < 27) bias = b_mod[ch - 18];
    v8f accA = {bias, bias, bias, bias, bias, bias, bias, bias};
    v8f accB = {};
    const _Float16* row = &P[(mt * 16 + m) * PROW];
    AV aA = load_afrag(row, 0, hi);
    AV aB = load_afrag(row, 9, hi);
    #pragma unroll
    for (int t = 0; t < 9; ++t) {
      AV aAn, aBn;
      int tn = (t < 8) ? t + 1 : t;     // prefetch next (dup on last iter)
      aAn = load_afrag(row, tn, hi);
      aBn = load_afrag(row, tn + 9, hi);
      v16h bmA = *(const v16h*)(wom + (((t * 2 + nt) * 32 + lane) << 4));
      v16h bmB = *(const v16h*)(wom + ((((t + 9) * 2 + nt) * 32 + lane) << 4));
      accA = __builtin_amdgcn_wmma_f32_16x16x32_f16(
          false, aA.v, false, bmA, (short)0, accA, false, false);
      accB = __builtin_amdgcn_wmma_f32_16x16x32_f16(
          false, aB.v, false, bmB, (short)0, accB, false, false);
      aA = aAn; aB = aBn;
    }
    v8f acc = accA + accB;
    if (ch < 27) {
      #pragma unroll
      for (int r = 0; r < 8; ++r) {
        int ml = mt * 16 + r + 8 * hi;
        float v = acc[r];
        OM[ml][ch] = (ch < 18) ? v : 2.0f / (1.0f + __expf(-v));
      }
    }
  }
  __syncthreads();

  // ---- Stage 3: modulated bilinear sampling, overwrite P ----
  for (int task = tid; task < TILE * KTAP; task += 256) {
    int i = task / KTAP, tap = task % KTAP;
    float dy = OM[i][2 * tap];
    float dx = OM[i][2 * tap + 1];
    float mo = OM[i][18 + tap];
    float py = (float)(y - 1 + tap / 3) + dy;
    float px = (float)(xb + i - 1 + tap % 3) + dx;
    float fy = floorf(py), fx = floorf(px);
    float wy = py - fy, wx = px - fx;
    int yi0 = (int)fy, xi0 = (int)fx;
    int yi1 = yi0 + 1, xi1 = xi0 + 1;
    float vy0 = (yi0 >= 0 && yi0 < Hn) ? 1.0f : 0.0f;
    float vy1 = (yi1 >= 0 && yi1 < Hn) ? 1.0f : 0.0f;
    float vx0 = (xi0 >= 0 && xi0 < Wn) ? 1.0f : 0.0f;
    float vx1 = (xi1 >= 0 && xi1 < Wn) ? 1.0f : 0.0f;
    float w00 = (1.0f - wy) * (1.0f - wx) * vy0 * vx0 * mo;
    float w01 = (1.0f - wy) * wx * vy0 * vx1 * mo;
    float w10 = wy * (1.0f - wx) * vy1 * vx0 * mo;
    float w11 = wy * wx * vy1 * vx1 * mo;
    int yc0 = min(max(yi0, 0), Hn - 1), yc1 = min(max(yi1, 0), Hn - 1);
    int xc0 = min(max(xi0, 0), Wn - 1), xc1 = min(max(xi1, 0), Wn - 1);
    const _Float16* r00 = xh + (((size_t)(b * Hn + yc0) * Wn + xc0) << 6);
    const _Float16* r01 = xh + (((size_t)(b * Hn + yc0) * Wn + xc1) << 6);
    const _Float16* r10 = xh + (((size_t)(b * Hn + yc1) * Wn + xc0) << 6);
    const _Float16* r11 = xh + (((size_t)(b * Hn + yc1) * Wn + xc1) << 6);
    h2* dst = (h2*)&P[i * PROW + tap * 64];
    #pragma unroll 4
    for (int c = 0; c < Cn; c += 2) {
      h2 a00 = *(const h2*)(r00 + c);
      h2 a01 = *(const h2*)(r01 + c);
      h2 a10 = *(const h2*)(r10 + c);
      h2 a11 = *(const h2*)(r11 + c);
      float s0 = w00 * (float)a00[0] + w01 * (float)a01[0] +
                 w10 * (float)a10[0] + w11 * (float)a11[0];
      float s1 = w00 * (float)a00[1] + w01 * (float)a01[1] +
                 w10 * (float)a10[1] + w11 * (float)a11[1];
      h2 o; o[0] = (_Float16)s0; o[1] = (_Float16)s1;
      dst[c >> 1] = o;
    }
  }
  __syncthreads();

  // ---- Stage 4: output GEMM (M=64,N=64,K=576) ----
  // 16 tiles = 8 waves x 2 independent m-tiles, A fragments double-buffered.
  {
    const int nt  = wave & 3;           // 0..3
    const int mt0 = (wave >> 2) << 1;   // 0 or 2
    const int o   = nt * 16 + m;
    v8f acc0 = {};
    v8f acc1 = {};
    const _Float16* row0 = &P[(mt0 * 16 + m) * PROW];
    const _Float16* row1 = &P[((mt0 + 1) * 16 + m) * PROW];
    AV a0 = load_afrag(row0, 0, hi);
    AV a1 = load_afrag(row1, 0, hi);
    #pragma unroll
    for (int t = 0; t < KSTEPS; ++t) {
      AV a0n, a1n;
      int tn = (t < KSTEPS - 1) ? t + 1 : t;
      a0n = load_afrag(row0, tn, hi);
      a1n = load_afrag(row1, tn, hi);
      v16h bm = *(const v16h*)(wrg + (((t * 4 + nt) * 32 + lane) << 4));
      acc0 = __builtin_amdgcn_wmma_f32_16x16x32_f16(
          false, a0.v, false, bm, (short)0, acc0, false, false);
      acc1 = __builtin_amdgcn_wmma_f32_16x16x32_f16(
          false, a1.v, false, bm, (short)0, acc1, false, false);
      a0 = a0n; a1 = a1n;
    }
    float* obase = out + (((size_t)b * On + o) << 14) + y * Wn + xb;
    #pragma unroll
    for (int r = 0; r < 8; ++r) {
      obase[mt0 * 16 + r + 8 * hi]        = acc0[r];
      obase[(mt0 + 1) * 16 + r + 8 * hi]  = acc1[r];
    }
  }
}

// ---------------------------------------------------------------------------
// inputs: 0:x 1:flows(unused) 2:w_off 3:b_off 4:w_mod 5:b_mod 6:w_reg
// ---------------------------------------------------------------------------
extern "C" void kernel_launch(void* const* d_in, const int* in_sizes, int n_in,
                              void* d_out, int out_size, void* d_ws, size_t ws_size,
                              hipStream_t stream) {
  const float* x     = (const float*)d_in[0];
  const float* w_off = (const float*)d_in[2];
  const float* b_off = (const float*)d_in[3];
  const float* w_mod = (const float*)d_in[4];
  const float* b_mod = (const float*)d_in[5];
  const float* w_reg = (const float*)d_in[6];
  float* out = (float*)d_out;

  char* ws = (char*)d_ws;
  _Float16* xh  = (_Float16*)ws;                                       // 16 MiB
  _Float16* wom = (_Float16*)(ws + (size_t)Bn * Hn * Wn * Cn * 2);     // 36 KiB
  _Float16* wrg = (_Float16*)(ws + (size_t)Bn * Hn * Wn * Cn * 2 + 18432 * 2);

  nhwc_kernel<<<Bn * 2 * 512, 256, 0, stream>>>(x, xh);
  pack_wom_kernel<<<(18432 + 255) / 256, 256, 0, stream>>>(w_off, w_mod, wom);
  pack_wreg_kernel<<<(36864 + 255) / 256, 256, 0, stream>>>(w_reg, wrg);
  dcn_main_kernel<<<(Bn * Hn * Wn) / TILE, 256, 0, stream>>>(
      xh, wom, wrg, b_off, b_mod, out);
}
